// MaskedAttentionHead_43817256353862
// MI455X (gfx1250) — matrix operand order repaired and checked by reading, not compiled, not run on hardware
//
#include <hip/hip_runtime.h>
#include <stdint.h>

typedef __attribute__((ext_vector_type(16))) _Float16 v16h;
typedef __attribute__((ext_vector_type(8)))  _Float16 v8h;
typedef __attribute__((ext_vector_type(8)))  float    v8f;
typedef __attribute__((ext_vector_type(4)))  unsigned int v4u;
typedef __attribute__((ext_vector_type(8)))  int      v8i;
typedef __attribute__((ext_vector_type(4)))  int      v4i;

#define D_MODEL 1024
#define D_K     64
#define SEQ     2048
#define NBATCH  4
#define NROWS   (NBATCH * SEQ)   // 8192

__device__ __forceinline__ v8f wmma_f16(v16h a, v16h b, v8f c) {
  // D = A(16x32 f16) * B(32x16 f16) + C(16x16 f32)
  return __builtin_amdgcn_wmma_f32_16x16x32_f16(
      /*neg_a=*/false, a, /*neg_b=*/false, b,
      /*c_mod=*/(short)0, c, /*reuse_a=*/false, /*reuse_b=*/false);
}

// XOR-butterfly max across the 16-lane row groups using DPP ROW_XMASK
// (pure VALU: v_mov_b32_dpp + v_max; no LDS traffic, no dscnt waits).
template <int CTRL>
__device__ __forceinline__ float dpp_xor_f32(float v) {
  return __int_as_float(__builtin_amdgcn_update_dpp(
      0, __float_as_int(v), CTRL, 0xf, 0xf, true));
}
__device__ __forceinline__ float rowmax16(float v) {
  v = fmaxf(v, dpp_xor_f32<0x161>(v));   // xor 1
  v = fmaxf(v, dpp_xor_f32<0x162>(v));   // xor 2
  v = fmaxf(v, dpp_xor_f32<0x164>(v));   // xor 4
  v = fmaxf(v, dpp_xor_f32<0x168>(v));   // xor 8
  return v;
}

// ---------------------------------------------------------------------------
// Tensor Data Mover: DMA a 2-D tile (f16 elements) from global to LDS.
// D# per CDNA5 ISA ch.8: group0 = {count, lds_addr, global_addr, type=2},
// group1 = {data_size=2B, tensor dims, tile dims, dim0 stride}.  Tracked
// with TENSORcnt; EXEC is ignored (issue from one wave only).
// This toolchain exposes the 6-arg builtin: (g0, g1, g2, g3, g4, cpol).
// ---------------------------------------------------------------------------
__device__ __forceinline__ void tdm_load_f16(uint32_t lds_addr, const void* gaddr,
                                             uint32_t tensor_d0, uint32_t tensor_d1,
                                             uint32_t tile_d0, uint32_t tile_d1,
                                             uint64_t stride0_elems) {
  const uint64_t ga = (uint64_t)(uintptr_t)gaddr;
  v4u g0;
  g0[0] = 1u;                                       // count=1 (valid user D#)
  g0[1] = lds_addr;                                 // LDS byte address
  g0[2] = (uint32_t)(ga & 0xFFFFFFFFu);             // global_addr[31:0]
  g0[3] = (uint32_t)((ga >> 32) & 0x01FFFFFFu)      // global_addr[56:32]
          | 0x80000000u;                            // type = 2 ("image")
  v8i g1;
  g1[0] = (int)(1u << 16);                          // data_size=1 (2 bytes)
  g1[1] = (int)((tensor_d0 & 0xFFFFu) << 16);                        // td0[15:0]
  g1[2] = (int)((tensor_d0 >> 16) | ((tensor_d1 & 0xFFFFu) << 16));  // td0[31:16]|td1[15:0]
  g1[3] = (int)((tensor_d1 >> 16) | ((tile_d0 & 0xFFFFu) << 16));    // td1[31:16]|tile0
  g1[4] = (int)(tile_d1 & 0xFFFFu);                 // tile_dim1 (tile_dim2=0)
  g1[5] = (int)(uint32_t)(stride0_elems & 0xFFFFFFFFu);   // dim0 stride lo
  g1[6] = (int)(uint32_t)((stride0_elems >> 32) & 0xFFFFu); // stride hi (dim1 stride=0)
  g1[7] = 0;
  const v4i zero4 = {0, 0, 0, 0};                   // groups 2/3 unused (<=2D)
  const v8i zero8 = {0, 0, 0, 0, 0, 0, 0, 0};
  __builtin_amdgcn_tensor_load_to_lds(g0, g1, zero4, zero4, zero8, 0);
}

// ---------------------------------------------------------------------------
// Projection: out[s, dk] = sum_d X[s, d] * W[dk, d]   (X:[8192,1024] f32,
// W:[64,1024] f32).  Each wave owns a 16-row tile and all 64 output columns
// (4 N-tiles), K-loop in steps of 32 with f16 WMMA / f32 accumulation.
// TRANSPOSE_OUT stores vh as [b][64][2048] so PV B-columns become contiguous.
// ---------------------------------------------------------------------------
template <bool TRANSPOSE_OUT>
__global__ void proj_kernel(const float* __restrict__ X,
                            const float* __restrict__ W,
                            _Float16* __restrict__ out) {
  const int lane   = threadIdx.x & 31;
  const int wave   = threadIdx.x >> 5;
  const int sbase  = (blockIdx.x * 4 + wave) * 16;  // global row base
  const int laneLo = lane & 15;
  const int hi     = lane >> 4;   // 0: lanes 0-15, 1: lanes 16-31
  const int c0     = hi * 8;      // unused; A loads also use bk (contiguous 16s)
  const int bk     = hi * 16;     // B-matrix K chunk base (contiguous 16s)

  v8f acc[4] = {};  // 4 N-tiles of 16 columns

  const float* xrow = X + (size_t)(sbase + laneLo) * D_MODEL;

  for (int k0 = 0; k0 < D_MODEL; k0 += 32) {
    v16h a;
#pragma unroll
    for (int j = 0; j < 8; ++j) {
      a[j]     = (_Float16)xrow[k0 + bk + j];
      a[8 + j] = (_Float16)xrow[k0 + bk + 8 + j];
    }
#pragma unroll
    for (int t = 0; t < 4; ++t) {
      const float* wrow = W + (size_t)(t * 16 + laneLo) * D_MODEL + k0 + bk;
      v16h bmat;
#pragma unroll
      for (int j = 0; j < 16; ++j) bmat[j] = (_Float16)wrow[j];
      acc[t] = wmma_f16(a, bmat, acc[t]);
    }
  }

#pragma unroll
  for (int t = 0; t < 4; ++t) {
#pragma unroll
    for (int r = 0; r < 8; ++r) {
      const int row = sbase + r + hi * 8;
      const int col = t * 16 + laneLo;
      const _Float16 v = (_Float16)acc[t][r];
      if (TRANSPOSE_OUT) {
        const int bidx = row >> 11;
        const int sl   = row & (SEQ - 1);
        out[((size_t)bidx * D_K + col) * SEQ + sl] = v;
      } else {
        out[(size_t)row * D_K + col] = v;
      }
    }
  }
}

// ---------------------------------------------------------------------------
// Flash-style attention.  4 waves/block, one 16-query tile per wave; all
// waves of a block share the same batch (blocks align with 128-tile batch
// stripes), so K/V blocks are staged once per block into double-buffered
// LDS by the Tensor Data Mover (issued by wave 0, TENSORcnt-tracked) and
// consumed by all 4 waves.  Row max via DPP row_xmask; row sums via a
// ones-matrix WMMA; P restaged C->A layout through per-wave LDS.
// ---------------------------------------------------------------------------
__global__ void attn_kernel(const _Float16* __restrict__ qh,
                            const _Float16* __restrict__ kh,
                            const _Float16* __restrict__ vhT,
                            const float* __restrict__ mmask,
                            float* __restrict__ out) {
  __shared__ __align__(16) _Float16 kbuf[2][32 * D_K];   // 2 x 4 KB
  __shared__ __align__(16) _Float16 vbuf[2][D_K * 32];   // 2 x 4 KB
  __shared__ __align__(16) _Float16 p_lds[4][16][32];    // per-wave P staging

  const int lane    = threadIdx.x & 31;
  const int wave    = threadIdx.x >> 5;
  const int tileIdx = blockIdx.x * 4 + wave;   // 0..511
  const int b       = tileIdx >> 7;            // constant per block
  const int qbase   = (tileIdx & 127) * 16;
  const int laneLo  = lane & 15;
  const int hi      = lane >> 4;
  const int c0      = hi * 8;                  // unused; A tiles also load from bk
  const int bk      = hi * 16;                 // B-layout chunk base

  const _Float16* qb  = qh  + (size_t)b * SEQ * D_K;
  const _Float16* kb_ = kh  + (size_t)b * SEQ * D_K;
  const _Float16* vb  = vhT + (size_t)b * D_K * SEQ;

  // q A-tiles (K = 0..31 and 32..63)
  v16h qA[2];
  {
    const _Float16* qrow = qb + (size_t)(qbase + laneLo) * D_K;
#pragma unroll
    for (int h = 0; h < 2; ++h) {
      v8h lo  = *(const v8h*)(qrow + h * 32 + bk);
      v8h hi8 = *(const v8h*)(qrow + h * 32 + bk + 8);
      qA[h] = __builtin_shufflevector(lo, hi8, 0, 1, 2, 3, 4, 5, 6, 7,
                                      8, 9, 10, 11, 12, 13, 14, 15);
    }
  }

  float msc[8];
#pragma unroll
  for (int r = 0; r < 8; ++r)
    msc[r] = 0.125f * mmask[(size_t)b * SEQ + qbase + r + hi * 8];

  v16h onesB;
#pragma unroll
  for (int j = 0; j < 16; ++j) onesB[j] = (_Float16)1.0f;

  v8f acc[4] = {};
  v8f lacc   = {};
  float mrun[8];
#pragma unroll
  for (int r = 0; r < 8; ++r) mrun[r] = -3.0e38f;

  // ---- prologue: stage key/value block 0 via TDM ----------------------
  if (wave == 0) {
    tdm_load_f16((uint32_t)(uintptr_t)&kbuf[0][0], kb_,
                 /*td0=*/32 * D_K, /*td1=*/1, /*tile0=*/32 * D_K, /*tile1=*/0,
                 /*stride0=*/32 * D_K);
    tdm_load_f16((uint32_t)(uintptr_t)&vbuf[0][0], vb,
                 /*td0=*/32, /*td1=*/D_K, /*tile0=*/32, /*tile1=*/D_K,
                 /*stride0=*/SEQ);
    __builtin_amdgcn_s_wait_tensorcnt(0);
  }
  __syncthreads();

  for (int kb0 = 0; kb0 < SEQ; kb0 += 32) {
    const int cur = (kb0 >> 5) & 1;
    const bool more = (kb0 + 32) < SEQ;

    // issue DMA for the next block into the other buffer (overlaps compute)
    if (wave == 0 && more) {
      tdm_load_f16((uint32_t)(uintptr_t)&kbuf[cur ^ 1][0],
                   kb_ + (size_t)(kb0 + 32) * D_K,
                   32 * D_K, 1, 32 * D_K, 0, 32 * D_K);
      tdm_load_f16((uint32_t)(uintptr_t)&vbuf[cur ^ 1][0],
                   vb + (kb0 + 32),
                   32, D_K, 32, D_K, SEQ);
    }

    const _Float16* kL = &kbuf[cur][0];
    const _Float16* vL = &vbuf[cur][0];

    // --- scores for the two 16-key sub-tiles (from LDS) ----------------
    v8f sc[2];
#pragma unroll
    for (int j2 = 0; j2 < 2; ++j2) {
      const _Float16* krow = kL + (j2 * 16 + laneLo) * D_K;
      v16h kB0 = *(const v16h*)(krow + bk);
      v16h kB1 = *(const v16h*)(krow + 32 + bk);
      v8f z = {};
      z = wmma_f16(qA[0], kB0, z);
      z = wmma_f16(qA[1], kB1, z);
      sc[j2] = z;
    }

    // --- online softmax: DPP row max, rescale, exp ---------------------
    float p0[8], p1[8];
#pragma unroll
    for (int r = 0; r < 8; ++r) {
      const float a0 = sc[0][r] * msc[r];
      const float a1 = sc[1][r] * msc[r];
      const float bmx  = rowmax16(fmaxf(a0, a1));
      const float mn   = fmaxf(mrun[r], bmx);
      const float corr = __expf(mrun[r] - mn);
      mrun[r] = mn;
      p0[r] = __expf(a0 - mn);
      p1[r] = __expf(a1 - mn);
      lacc[r]   *= corr;
      acc[0][r] *= corr; acc[1][r] *= corr;
      acc[2][r] *= corr; acc[3][r] *= corr;
    }

    // --- restage P (C layout -> row-major LDS -> A layout) -------------
#pragma unroll
    for (int r = 0; r < 8; ++r) {
      p_lds[wave][r + hi * 8][laneLo]      = (_Float16)p0[r];
      p_lds[wave][r + hi * 8][16 + laneLo] = (_Float16)p1[r];
    }
    v16h pA;
    {
      const _Float16* prow = &p_lds[wave][laneLo][0];
      v8h lo  = *(const v8h*)(prow + bk);
      v8h hi8 = *(const v8h*)(prow + bk + 8);
      pA = __builtin_shufflevector(lo, hi8, 0, 1, 2, 3, 4, 5, 6, 7,
                                   8, 9, 10, 11, 12, 13, 14, 15);
    }

    // --- row sums via ones-matrix WMMA ---------------------------------
    lacc = wmma_f16(pA, onesB, lacc);

    // --- PV: acc[t] += P(16x32) * V(32x16 per N-tile) (from LDS) -------
#pragma unroll
    for (int t = 0; t < 4; ++t) {
      const _Float16* vcol = vL + (t * 16 + laneLo) * 32;
      v16h vB = *(const v16h*)(vcol + bk);
      acc[t] = wmma_f16(pA, vB, acc[t]);
    }

    // next buffer ready + everyone done with this one before reuse
    if (wave == 0 && more) __builtin_amdgcn_s_wait_tensorcnt(0);
    __syncthreads();
  }

  // --- normalize and store ---------------------------------------------
#pragma unroll
  for (int t = 0; t < 4; ++t)
#pragma unroll
    for (int r = 0; r < 8; ++r) {
      const int row = qbase + r + hi * 8;
      const int col = t * 16 + laneLo;
      out[((size_t)b * SEQ + row) * D_K + col] = acc[t][r] / lacc[r];
    }
}

// ---------------------------------------------------------------------------
extern "C" void kernel_launch(void* const* d_in, const int* in_sizes, int n_in,
                              void* d_out, int out_size, void* d_ws, size_t ws_size,
                              hipStream_t stream) {
  const float* q  = (const float*)d_in[0];
  const float* k  = (const float*)d_in[1];
  const float* v  = (const float*)d_in[2];
  const float* m  = (const float*)d_in[3];
  const float* Wq = (const float*)d_in[4];
  const float* Wk = (const float*)d_in[5];
  const float* Wv = (const float*)d_in[6];
  float* out = (float*)d_out;

  // Workspace: qh | kh | vhT, each 8192*64 f16 = 1 MB (3 MB total)
  char* ws = (char*)d_ws;
  const size_t projBytes = (size_t)NROWS * D_K * sizeof(_Float16);
  _Float16* qh  = (_Float16*)(ws);
  _Float16* kh  = (_Float16*)(ws + projBytes);
  _Float16* vhT = (_Float16*)(ws + 2 * projBytes);

  dim3 blk(128);
  dim3 grd(NROWS / 64);
  proj_kernel<false><<<grd, blk, 0, stream>>>(q, Wq, qh);
  proj_kernel<false><<<grd, blk, 0, stream>>>(k, Wk, kh);
  proj_kernel<true ><<<grd, blk, 0, stream>>>(v, Wv, vhT);

  attn_kernel<<<dim3(128), dim3(128), 0, stream>>>(qh, kh, vhT, m, out);
}
